// MoETransformerBlock_22789096472718
// MI455X (gfx1250) — matrix operand order repaired and checked by reading, not compiled, not run on hardware
//
#include <hip/hip_runtime.h>
#include <math.h>

// ---- fixed problem shape ----
#define CB   2
#define CT   2048
#define CDM  1024
#define CH   16
#define CDK  64
#define CDFF 4096
#define CNE  8
#define CBT  (CB*CT)   // 4096 tokens

typedef __bf16 bf16x16 __attribute__((ext_vector_type(16)));
typedef float  v8f     __attribute__((ext_vector_type(8)));

struct alignas(16) U128 { unsigned int a, b, c, d; };

union Frag16 {            // 16 bf16 = 32 bytes = one WMMA A/B operand
  bf16x16 v;
  U128 q[2];
  unsigned short s[16];
};

__device__ __forceinline__ unsigned short f2bf(float f) {
  unsigned int u = __float_as_uint(f);
  u += 0x7fffu + ((u >> 16) & 1u);      // round-to-nearest-even
  return (unsigned short)(u >> 16);
}

// one 16-byte async global->LDS copy (CDNA5, tracked by ASYNCcnt)
__device__ __forceinline__ void async_copy16(const unsigned short* g, unsigned short* l) {
  unsigned lds_off = (unsigned)(uintptr_t)l;            // LDS aperture: offset in addr[31:0]
  unsigned long long ga = (unsigned long long)(uintptr_t)g;
  asm volatile("global_load_async_to_lds_b128 %0, %1, off"
               :: "v"(lds_off), "v"(ga) : "memory");
}
__device__ __forceinline__ void wait_async_all() {
  asm volatile("s_wait_asynccnt 0x0" ::: "memory");
}

// ---------------- weight f32 -> bf16 with transpose ----------------
// in: [K][N] f32 row-major (per expert slice via blockIdx.z), out: [N][K] bf16
__global__ __launch_bounds__(256)
void cvt_bf16_t(const float* __restrict__ in, unsigned short* __restrict__ out, int K, int N) {
  __shared__ unsigned short tile[32][33];
  in  += (size_t)blockIdx.z * K * N;
  out += (size_t)blockIdx.z * K * N;
  const int k0 = blockIdx.y * 32, n0 = blockIdx.x * 32;
  const int tx = threadIdx.x & 31, ty = threadIdx.x >> 5;   // 8 rows of 32
#pragma unroll
  for (int i = ty; i < 32; i += 8)
    tile[i][tx] = f2bf(in[(size_t)(k0 + i) * N + n0 + tx]);
  __syncthreads();
#pragma unroll
  for (int i = ty; i < 32; i += 8)
    out[(size_t)(n0 + i) * K + k0 + tx] = tile[tx][i];
}

// ---------------- RMSNorm over D=1024 ----------------
__global__ __launch_bounds__(256)
void rmsnorm_k(const float* __restrict__ x, const float* __restrict__ w,
               unsigned short* __restrict__ obf, float* __restrict__ of) {
  __shared__ float red[256];
  const int row = blockIdx.x, tid = threadIdx.x;
  const float* xr = x + (size_t)row * CDM;
  float ss = 0.f;
  for (int i = tid; i < CDM; i += 256) { float v = xr[i]; ss += v * v; }
  red[tid] = ss; __syncthreads();
  for (int s = 128; s > 0; s >>= 1) { if (tid < s) red[tid] += red[tid + s]; __syncthreads(); }
  const float scale = rsqrtf(red[0] / (float)CDM + 1e-6f);
  for (int i = tid; i < CDM; i += 256) {
    float v = xr[i] * scale * w[i];
    obf[(size_t)row * CDM + i] = f2bf(v);
    if (of) of[(size_t)row * CDM + i] = v;
  }
}

// ---------------- generic bf16 GEMM, 128x128 tile, async double-buffer ----------------
// A: [M][K] bf16 row-major (optionally gathered via rowA).
// Bt: [N][K] bf16 (pre-transposed weights).
// cntp: device row count (compacted MoE); blocks fully above it exit early.
enum { EP_PLAIN = 0, EP_RES = 1, EP_GELU_BF = 2, EP_GATED_ACC = 3 };

__global__ __launch_bounds__(256)
void gemm_bf16(const unsigned short* __restrict__ A,
               const unsigned short* __restrict__ Bt,
               int M, int N, int K,
               const float* __restrict__ bias,
               const float* __restrict__ residual,
               const float* __restrict__ gates, int gate_e,
               float* __restrict__ outF,
               unsigned short* __restrict__ outBF,
               int mode,
               const int* __restrict__ cntp,      // null -> all M rows valid
               const int* __restrict__ rowA,      // null -> identity gather
               const int* __restrict__ rowOut) {  // null -> identity scatter
  const int blockRow = blockIdx.y * 128;
  const int blockCol = blockIdx.x * 128;
  const int Mv = cntp ? cntp[0] : M;
  if (blockRow >= Mv) return;          // dead MoE block: retire before any barrier

  __shared__ unsigned short lds_a[2][128 * 64];   // [row][k], K-slab = 64
  __shared__ unsigned short lds_b[2][128 * 64];   // [col][k]
  const int tid  = threadIdx.x;
  const int lane = tid & 31;
  const int wid  = tid >> 5;
  const int ln   = lane & 15;
  const int hi   = lane >> 4;
  const int waveM = (wid & 1) * 64;
  const int waveN = (wid >> 1) * 32;

  // resolve this thread's 4 A-source rows once (gather indirection for MoE)
  int srcRow[4];
#pragma unroll
  for (int c = 0; c < 4; ++c) {
    int gr = blockRow + (((c * 256 + tid)) >> 3);
    srcRow[c] = rowA ? rowA[gr] : gr;
  }

  v8f acc[4][2];
#pragma unroll
  for (int mt = 0; mt < 4; ++mt)
#pragma unroll
    for (int nt = 0; nt < 2; ++nt)
#pragma unroll
      for (int r = 0; r < 8; ++r) acc[mt][nt][r] = 0.f;

  // stage one 128x64 A slab + 128x64 B slab asynchronously into buffer `buf`
  auto stage = [&](int k0, int buf) {
#pragma unroll
    for (int c = 0; c < 4; ++c) {
      int idx = c * 256 + tid;
      int r = idx >> 3, seg = (idx & 7) * 8;
      async_copy16(A  + (size_t)srcRow[c] * K + k0 + seg, &lds_a[buf][r * 64 + seg]);
      async_copy16(Bt + (size_t)(blockCol + r) * K + k0 + seg, &lds_b[buf][r * 64 + seg]);
    }
  };

  stage(0, 0);
  wait_async_all();
  __syncthreads();

  const int nIter = K >> 6;   // K/64
  for (int it = 0; it < nIter; ++it) {
    const int cur = it & 1;
    if (it + 1 < nIter) stage((it + 1) << 6, cur ^ 1);

#pragma unroll
    for (int kk = 0; kk < 64; kk += 32) {
      Frag16 a[4];
#pragma unroll
      for (int mt = 0; mt < 4; ++mt) {
        const unsigned short* p = &lds_a[cur][(waveM + mt * 16 + ln) * 64 + kk + hi * 8];
        a[mt].q[0] = *(const U128*)p;          // K = base+0..7
        a[mt].q[1] = *(const U128*)(p + 16);   // K = base+16..23
      }
      Frag16 b[2];
#pragma unroll
      for (int nt = 0; nt < 2; ++nt) {
        const unsigned short* p = &lds_b[cur][(waveN + nt * 16 + ln) * 64 + kk + hi * 8];
        b[nt].q[0] = *(const U128*)p;          // K = base+0..7
        b[nt].q[1] = *(const U128*)(p + 16);   // K = base+16..23
      }
#pragma unroll
      for (int mt = 0; mt < 4; ++mt)
#pragma unroll
        for (int nt = 0; nt < 2; ++nt)
          acc[mt][nt] = __builtin_amdgcn_wmma_f32_16x16x32_bf16(
              false, a[mt].v, false, b[nt].v, (short)0, acc[mt][nt], false, false);
    }
    wait_async_all();
    __syncthreads();
  }

  // epilogue: C layout M = r + 8*hi, N = ln
#pragma unroll
  for (int mt = 0; mt < 4; ++mt)
#pragma unroll
    for (int nt = 0; nt < 2; ++nt)
#pragma unroll
      for (int r = 0; r < 8; ++r) {
        int grow = blockRow + waveM + mt * 16 + hi * 8 + r;
        int gcol = blockCol + waveN + nt * 16 + ln;
        float val = acc[mt][nt][r];
        if (bias) val += bias[gcol];
        size_t o = (size_t)grow * N + gcol;
        if (mode == EP_PLAIN) {
          outF[o] = val;
        } else if (mode == EP_RES) {
          outF[o] = residual[o] + val;
        } else if (mode == EP_GELU_BF) {
          float g = 0.5f * val * (1.f + tanhf(0.7978845608028654f * (val + 0.044715f * val * val * val)));
          outBF[o] = f2bf(g);
        } else {  // EP_GATED_ACC: scatter-accumulate gated expert output per token
          if (grow < Mv) {
            int token = rowOut ? rowOut[grow] : grow;
            float gsc = gates[(size_t)token * CNE + gate_e];
            outF[(size_t)token * N + gcol] += gsc * val;
          }
        }
      }
}

// ---------------- fused q/k RMSNorm + RoPE + layout ----------------
// qkv: [token][3*1024] f32.  Outputs bf16: Q,K = [b*H+h][T][64], VT = [b*H+h][64][T]
__global__ __launch_bounds__(256)
void rope_qk(const float* __restrict__ qkv, const float* __restrict__ qn, const float* __restrict__ kn,
             unsigned short* __restrict__ Q, unsigned short* __restrict__ Kk,
             unsigned short* __restrict__ VT) {
  const int wid = threadIdx.x >> 5, lane = threadIdx.x & 31;
  const int gw = blockIdx.x * 8 + wid;   // token*H + h
  const int token = gw / CH, h = gw % CH;
  const int b = token / CT, pos = token % CT;
  const float* base = qkv + (size_t)token * (3 * CDM) + h * CDK;
  float q1 = base[2 * lane], q2 = base[2 * lane + 1];
  float k1 = base[CDM + 2 * lane], k2 = base[CDM + 2 * lane + 1];
  float v1 = base[2 * CDM + 2 * lane], v2 = base[2 * CDM + 2 * lane + 1];
  float ssq = q1 * q1 + q2 * q2, ssk = k1 * k1 + k2 * k2;
#pragma unroll
  for (int off = 16; off > 0; off >>= 1) {
    ssq += __shfl_xor(ssq, off, 32);
    ssk += __shfl_xor(ssk, off, 32);
  }
  float rq = rsqrtf(ssq / (float)CDK + 1e-6f);
  float rk = rsqrtf(ssk / (float)CDK + 1e-6f);
  q1 *= rq * qn[2 * lane]; q2 *= rq * qn[2 * lane + 1];
  k1 *= rk * kn[2 * lane]; k2 *= rk * kn[2 * lane + 1];
  float freq = __expf(-((float)(2 * lane) / (float)CDK) * 9.210340371976184f);  // ln(10000)
  float ang = (float)pos * freq, sn, cs;
  sincosf(ang, &sn, &cs);
  float qo1 = q1 * cs - q2 * sn, qo2 = q1 * sn + q2 * cs;
  float ko1 = k1 * cs - k2 * sn, ko2 = k1 * sn + k2 * cs;
  size_t bh = (size_t)(b * CH + h);
  size_t oq = (bh * CT + pos) * CDK + 2 * lane;
  Q[oq] = f2bf(qo1);  Q[oq + 1] = f2bf(qo2);
  Kk[oq] = f2bf(ko1); Kk[oq + 1] = f2bf(ko2);
  size_t ov = (bh * CDK + 2 * lane) * CT + pos;
  VT[ov] = f2bf(v1);  VT[ov + CT] = f2bf(v2);
}

// ---------------- flash attention: 1 wave = 16 q rows ----------------
__global__ __launch_bounds__(128)
void attn_k(const unsigned short* __restrict__ Q, const unsigned short* __restrict__ Kk,
            const unsigned short* __restrict__ VT, unsigned short* __restrict__ attn) {
  __shared__ unsigned short lds_p[4][16 * 32];   // per-wave P staging
  const int tid = threadIdx.x, lane = tid & 31, wid = tid >> 5;
  const int ln = lane & 15, hi = lane >> 4;
  const int bh = blockIdx.y, b = bh / CH, h = bh % CH;
  const int qbase = blockIdx.x * 64 + wid * 16;

  const unsigned short* qp = Q  + (size_t)bh * CT * CDK;
  const unsigned short* kp = Kk + (size_t)bh * CT * CDK;
  const unsigned short* vp = VT + (size_t)bh * CDK * CT;

  Frag16 aq[2];
#pragma unroll
  for (int kk = 0; kk < 2; ++kk) {
    const unsigned short* p = qp + (size_t)(qbase + ln) * CDK + kk * 32 + hi * 8;
    aq[kk].q[0] = *(const U128*)p;
    aq[kk].q[1] = *(const U128*)(p + 16);
  }

  float mrow[8], lrow[8];
  v8f accO[4];
#pragma unroll
  for (int r = 0; r < 8; ++r) { mrow[r] = -3.0e38f; lrow[r] = 0.f; }
#pragma unroll
  for (int d = 0; d < 4; ++d)
#pragma unroll
    for (int r = 0; r < 8; ++r) accO[d][r] = 0.f;

  const int ktmax = (qbase + 15) >> 5;
  for (int kt = 0; kt <= ktmax; ++kt) {
    const int tb = kt * 32;
    v8f S[2];
#pragma unroll
    for (int ns = 0; ns < 2; ++ns)
#pragma unroll
      for (int r = 0; r < 8; ++r) S[ns][r] = 0.f;

#pragma unroll
    for (int ns = 0; ns < 2; ++ns)
#pragma unroll
      for (int kk = 0; kk < 2; ++kk) {
        Frag16 bk;
        const unsigned short* p = kp + (size_t)(tb + ns * 16 + ln) * CDK + kk * 32 + hi * 8;
        bk.q[0] = *(const U128*)p;
        bk.q[1] = *(const U128*)(p + 16);
        S[ns] = __builtin_amdgcn_wmma_f32_16x16x32_bf16(
            false, aq[kk].v, false, bk.v, (short)0, S[ns], false, false);
      }

    // scale + causal mask (C layout: row = qbase+8*hi+r, col = tb+16*ns+ln)
#pragma unroll
    for (int ns = 0; ns < 2; ++ns)
#pragma unroll
      for (int r = 0; r < 8; ++r) {
        float s = S[ns][r] * 0.125f;
        int qpos = qbase + hi * 8 + r;
        int kpos = tb + ns * 16 + ln;
        S[ns][r] = (kpos <= qpos) ? s : -3.0e38f;
      }

    float alpha[8];
#pragma unroll
    for (int r = 0; r < 8; ++r) {
      float v = fmaxf(S[0][r], S[1][r]);
#pragma unroll
      for (int off = 8; off > 0; off >>= 1) v = fmaxf(v, __shfl_xor(v, off, 32));
      float mn = fmaxf(mrow[r], v);
      alpha[r] = __expf(mrow[r] - mn);
      float p0 = __expf(S[0][r] - mn);
      float p1 = __expf(S[1][r] - mn);
      float rs = p0 + p1;
#pragma unroll
      for (int off = 8; off > 0; off >>= 1) rs += __shfl_xor(rs, off, 32);
      lrow[r] = lrow[r] * alpha[r] + rs;
      mrow[r] = mn;
      S[0][r] = p0; S[1][r] = p1;
    }
#pragma unroll
    for (int d = 0; d < 4; ++d)
#pragma unroll
      for (int r = 0; r < 8; ++r) accO[d][r] *= alpha[r];

    // C layout -> A layout via wave-private LDS round trip
#pragma unroll
    for (int ns = 0; ns < 2; ++ns)
#pragma unroll
      for (int r = 0; r < 8; ++r)
        lds_p[wid][(hi * 8 + r) * 32 + ns * 16 + ln] = f2bf(S[ns][r]);
    asm volatile("s_wait_dscnt 0x0" ::: "memory");
    Frag16 ap;
    {
      const unsigned short* p = &lds_p[wid][ln * 32 + hi * 8];
      ap.q[0] = *(const U128*)p;
      ap.q[1] = *(const U128*)(p + 16);
    }
#pragma unroll
    for (int d = 0; d < 4; ++d) {
      Frag16 bv;
      const unsigned short* p = vp + (size_t)(d * 16 + ln) * CT + tb + hi * 8;
      bv.q[0] = *(const U128*)p;
      bv.q[1] = *(const U128*)(p + 16);
      accO[d] = __builtin_amdgcn_wmma_f32_16x16x32_bf16(
          false, ap.v, false, bv.v, (short)0, accO[d], false, false);
    }
  }

#pragma unroll
  for (int r = 0; r < 8; ++r) lrow[r] = 1.0f / lrow[r];
#pragma unroll
  for (int d = 0; d < 4; ++d)
#pragma unroll
    for (int r = 0; r < 8; ++r) {
      int t = qbase + hi * 8 + r;
      int dd = d * 16 + ln;
      attn[((size_t)(b * CT + t) * CDM) + h * CDK + dd] = f2bf(accO[d][r] * lrow[r]);
    }
}

// ---------------- MoE token compaction helpers ----------------
__global__ void zero_counts(int* __restrict__ c) {
  if (threadIdx.x < CNE) c[threadIdx.x] = 0;
}
__global__ __launch_bounds__(256)
void pad_lists(int* __restrict__ lists, const int* __restrict__ counts) {
  const int e = blockIdx.y;
  const int i = blockIdx.x * 256 + threadIdx.x;
  if (i < CBT && i >= counts[e]) lists[(size_t)e * CBT + i] = 0;
}

// ---------------- router: 1 wave per token ----------------
__global__ __launch_bounds__(256)
void router_k(const float* __restrict__ h2, const float* __restrict__ rw, const float* __restrict__ rb,
              float* __restrict__ probs, float* __restrict__ sel, float* __restrict__ gates,
              int* __restrict__ counts, int* __restrict__ lists) {
  const int wid = threadIdx.x >> 5, lane = threadIdx.x & 31;
  const int token = blockIdx.x * 8 + wid;
  const float* hr = h2 + (size_t)token * CDM;
  float acc[CNE];
#pragma unroll
  for (int e = 0; e < CNE; ++e) acc[e] = 0.f;
  for (int d = lane; d < CDM; d += 32) {
    float hv = hr[d];
#pragma unroll
    for (int e = 0; e < CNE; ++e) acc[e] += hv * rw[d * CNE + e];
  }
#pragma unroll
  for (int e = 0; e < CNE; ++e)
#pragma unroll
    for (int off = 16; off > 0; off >>= 1) acc[e] += __shfl_xor(acc[e], off, 32);
  if (lane == 0) {
    float mx = -3.0e38f;
#pragma unroll
    for (int e = 0; e < CNE; ++e) { acc[e] += rb[e]; mx = fmaxf(mx, acc[e]); }
    float s = 0.f, p[CNE];
#pragma unroll
    for (int e = 0; e < CNE; ++e) { p[e] = __expf(acc[e] - mx); s += p[e]; }
#pragma unroll
    for (int e = 0; e < CNE; ++e) { p[e] /= s; probs[(size_t)token * CNE + e] = p[e]; }
    int i1 = 0;
    for (int e = 1; e < CNE; ++e) if (p[e] > p[i1]) i1 = e;
    int i2 = (i1 == 0) ? 1 : 0;
    for (int e = 0; e < CNE; ++e) if (e != i1 && p[e] > p[i2]) i2 = e;
    float denom = p[i1] + p[i2];
#pragma unroll
    for (int e = 0; e < CNE; ++e) {
      float g = (e == i1) ? p[i1] / denom : (e == i2) ? p[i2] / denom : 0.f;
      gates[(size_t)token * CNE + e] = g;
      sel[(size_t)token * CNE + e]   = (e == i1 || e == i2) ? 1.f : 0.f;
    }
    int idx1 = atomicAdd(&counts[i1], 1);
    lists[(size_t)i1 * CBT + idx1] = token;
    int idx2 = atomicAdd(&counts[i2], 1);
    lists[(size_t)i2 * CBT + idx2] = token;
  }
}

// ---------------- deterministic aux-loss reduction ----------------
__global__ __launch_bounds__(256)
void aux_k(const float* __restrict__ probs, const float* __restrict__ sel, float* __restrict__ outp) {
  __shared__ float rp[256][CNE];
  __shared__ float rs[256][CNE];
  const int tid = threadIdx.x;
  float ap[CNE], as[CNE];
#pragma unroll
  for (int e = 0; e < CNE; ++e) { ap[e] = 0.f; as[e] = 0.f; }
  for (int t = tid; t < CBT; t += 256)
#pragma unroll
    for (int e = 0; e < CNE; ++e) {
      ap[e] += probs[(size_t)t * CNE + e];
      as[e] += sel[(size_t)t * CNE + e];
    }
#pragma unroll
  for (int e = 0; e < CNE; ++e) { rp[tid][e] = ap[e]; rs[tid][e] = as[e]; }
  __syncthreads();
  for (int s = 128; s > 0; s >>= 1) {
    if (tid < s)
#pragma unroll
      for (int e = 0; e < CNE; ++e) { rp[tid][e] += rp[tid + s][e]; rs[tid][e] += rs[tid + s][e]; }
    __syncthreads();
  }
  if (tid == 0) {
    float aux = 0.f;
#pragma unroll
    for (int e = 0; e < CNE; ++e)
      aux += (rs[0][e] / (float)CBT) * (rp[0][e] / (float)CBT);
    outp[0] = (float)CNE * aux;
  }
}

// ---------------- host orchestration ----------------
extern "C" void kernel_launch(void* const* d_in, const int* in_sizes, int n_in,
                              void* d_out, int out_size, void* d_ws, size_t ws_size,
                              hipStream_t stream) {
  (void)in_sizes; (void)n_in; (void)out_size; (void)ws_size;
  const float* x        = (const float*)d_in[0];
  const float* norm1_w  = (const float*)d_in[1];
  const float* qkv_w    = (const float*)d_in[2];
  const float* qn_w     = (const float*)d_in[3];
  const float* kn_w     = (const float*)d_in[4];
  const float* wo_w     = (const float*)d_in[5];
  const float* norm2_w  = (const float*)d_in[6];
  const float* router_w = (const float*)d_in[7];
  const float* router_b = (const float*)d_in[8];
  const float* w1       = (const float*)d_in[9];
  const float* b1       = (const float*)d_in[10];
  const float* w2       = (const float*)d_in[11];
  const float* b2       = (const float*)d_in[12];
  float* out = (float*)d_out;

  char* ws = (char*)d_ws;
  size_t off = 0;
  auto alloc = [&](size_t bytes) -> char* {
    char* p = ws + off;
    off += (bytes + 255) & ~(size_t)255;
    return p;
  };
  // all weights stored transposed: [N][K] bf16
  unsigned short* qkvw_bf = (unsigned short*)alloc((size_t)CDM * 3 * CDM * 2);
  unsigned short* wo_bf   = (unsigned short*)alloc((size_t)CDM * CDM * 2);
  unsigned short* w1_bf   = (unsigned short*)alloc((size_t)CNE * CDM * CDFF * 2);
  unsigned short* w2_bf   = (unsigned short*)alloc((size_t)CNE * CDFF * CDM * 2);
  unsigned short* h_bf    = (unsigned short*)alloc((size_t)CBT * CDM * 2);
  float*          qkv_f   = (float*)alloc((size_t)CBT * 3 * CDM * 4);
  unsigned short* q_bf    = (unsigned short*)alloc((size_t)CBT * CDM * 2);
  unsigned short* k_bf    = (unsigned short*)alloc((size_t)CBT * CDM * 2);
  unsigned short* vT_bf   = (unsigned short*)alloc((size_t)CBT * CDM * 2);
  unsigned short* attn_bf = (unsigned short*)alloc((size_t)CBT * CDM * 2);
  float*          xbuf    = (float*)alloc((size_t)CBT * CDM * 4);
  unsigned short* h2_bf   = (unsigned short*)alloc((size_t)CBT * CDM * 2);
  float*          h2_f    = (float*)alloc((size_t)CBT * CDM * 4);
  unsigned short* t1_bf   = (unsigned short*)alloc((size_t)CBT * CDFF * 2);
  float*          probs   = (float*)alloc((size_t)CBT * CNE * 4);
  float*          sel     = (float*)alloc((size_t)CBT * CNE * 4);
  float*          gates   = (float*)alloc((size_t)CBT * CNE * 4);
  int*            counts  = (int*)alloc((size_t)CNE * 4);
  int*            lists   = (int*)alloc((size_t)CNE * CBT * 4);

  // weights -> bf16, transposed to [N][K]
  cvt_bf16_t<<<dim3(3 * CDM / 32, CDM / 32, 1), 256, 0, stream>>>(qkv_w, qkvw_bf, CDM, 3 * CDM);
  cvt_bf16_t<<<dim3(CDM / 32, CDM / 32, 1), 256, 0, stream>>>(wo_w, wo_bf, CDM, CDM);
  cvt_bf16_t<<<dim3(CDFF / 32, CDM / 32, CNE), 256, 0, stream>>>(w1, w1_bf, CDM, CDFF);
  cvt_bf16_t<<<dim3(CDM / 32, CDFF / 32, CNE), 256, 0, stream>>>(w2, w2_bf, CDFF, CDM);

  // h = rmsnorm(x)
  rmsnorm_k<<<CBT, 256, 0, stream>>>(x, norm1_w, h_bf, nullptr);
  // qkv = h @ qkv_w
  gemm_bf16<<<dim3(3 * CDM / 128, CBT / 128), 256, 0, stream>>>(
      h_bf, qkvw_bf, CBT, 3 * CDM, CDM, nullptr, nullptr, nullptr, 0, qkv_f, nullptr,
      EP_PLAIN, nullptr, nullptr, nullptr);
  // q/k norm + rope + layout
  rope_qk<<<CBT * CH / 8, 256, 0, stream>>>(qkv_f, qn_w, kn_w, q_bf, k_bf, vT_bf);
  // attention
  attn_k<<<dim3(CT / 64, CB * CH), 128, 0, stream>>>(q_bf, k_bf, vT_bf, attn_bf);
  // x = x + attn @ wo
  gemm_bf16<<<dim3(CDM / 128, CBT / 128), 256, 0, stream>>>(
      attn_bf, wo_bf, CBT, CDM, CDM, nullptr, x, nullptr, 0, xbuf, nullptr,
      EP_RES, nullptr, nullptr, nullptr);
  // h2 = rmsnorm(x)
  rmsnorm_k<<<CBT, 256, 0, stream>>>(xbuf, norm2_w, h2_bf, h2_f);
  // router (+ top-2 token compaction) + aux
  zero_counts<<<1, 32, 0, stream>>>(counts);
  router_k<<<CBT / 8, 256, 0, stream>>>(h2_f, router_w, router_b, probs, sel, gates, counts, lists);
  pad_lists<<<dim3(CBT / 256, CNE), 256, 0, stream>>>(lists, counts);
  aux_k<<<1, 256, 0, stream>>>(probs, sel, out + (size_t)CBT * CDM);
  // out = x (then accumulate gated experts)
  hipMemcpyAsync(out, xbuf, (size_t)CBT * CDM * sizeof(float), hipMemcpyDeviceToDevice, stream);
  // compacted MoE: per expert, gathered w1-GEMM (gelu) then w2-GEMM scattered+gated
  for (int e = 0; e < CNE; ++e) {
    gemm_bf16<<<dim3(CDFF / 128, CBT / 128), 256, 0, stream>>>(
        h2_bf, w1_bf + (size_t)e * CDM * CDFF, CBT, CDFF, CDM,
        b1 + (size_t)e * CDFF, nullptr, nullptr, 0, nullptr, t1_bf,
        EP_GELU_BF, counts + e, lists + (size_t)e * CBT, nullptr);
    gemm_bf16<<<dim3(CDM / 128, CBT / 128), 256, 0, stream>>>(
        t1_bf, w2_bf + (size_t)e * CDFF * CDM, CBT, CDM, CDFF,
        b2 + (size_t)e * CDM, nullptr, gates, e, out, nullptr,
        EP_GATED_ACC, counts + e, nullptr, lists + (size_t)e * CBT);
  }
}
